// VQ_AE_28389733826788
// MI455X (gfx1250) — matrix operand order repaired and checked by reading, not compiled, not run on hardware
//
#include <hip/hip_runtime.h>

// ---------------------------------------------------------------------------
// 3D VQ-VAE forward for MI455X (gfx1250, wave32).
// All GEMMs run on v_wmma_f32_16x16x32_bf16 (bf16 inputs, f32 accumulate).
// GEMM kernel contract (guaranteed by every launch below):
//   M % 64 == 0, K % 32 == 0.  N arbitrary (out-of-range columns are
//   address-clamped on load and masked on store).
// Pipeline: register-prefetch of next K-tile + double-buffered LDS
// (one workgroup barrier per K-step; per-wave dscnt wait before WMMA makes
// the 2-deep buffer reuse safe), global_prefetch at distance 2.
// fp32->bf16 uses hardware converts (v_cvt_pk_bf16_f32, RTNE).
// ---------------------------------------------------------------------------

#define DEVINL __device__ __forceinline__

typedef __attribute__((ext_vector_type(16))) __bf16        bf16x16;
typedef __attribute__((ext_vector_type(2)))  __bf16        bf16x2;
typedef __attribute__((ext_vector_type(8)))  float         f32x8;
typedef __attribute__((ext_vector_type(4)))  unsigned int  u32x4;

// ---------------- helpers ----------------

DEVINL unsigned short f2bf(float f) {
  // native fptrunc f32 -> bf16 (RTNE on gfx1250)
  return __builtin_bit_cast(unsigned short, (__bf16)f);
}
DEVINL unsigned int pack2(float lo, float hi) {
  bf16x2 t;
  t.x = (__bf16)lo;
  t.y = (__bf16)hi;
  return __builtin_bit_cast(unsigned int, t);
}

DEVINL f32x8 wmma_bf16(bf16x16 a, bf16x16 b, f32x8 c) {
  // D = A(16x32) * B(32x16) + C(16x16), f32 accumulate
  return __builtin_amdgcn_wmma_f32_16x16x32_bf16(
      /*neg_a=*/false, a, /*neg_b=*/false, b,
      /*c_mod=*/(short)0, c, /*reuse_a=*/false, /*reuse_b=*/false);
}

// LDS pitch: 40 bf16 (= 20 dwords = 80 B) -> 16B-aligned fragment chunks,
// conflict-free bank stride for both A-row and B-column reads.
#define LDS_PITCH_U32 20

// A fragment (16x32 bf16, ISA 7.12.2 layout).
DEVINL bf16x16 fragA(const unsigned int* s, int rowBase, int lane) {
  int m  = lane & 15;
  int kh = (lane >> 4) << 2;              // dword offset: 0 or 4
  const unsigned int* p = s + (rowBase + m) * LDS_PITCH_U32 + kh;
  union { u32x4 q[2]; bf16x16 v; } u;
  u.q[0] = *(const u32x4*)(p);            // K 0..7   / 8..15
  u.q[1] = *(const u32x4*)(p + 8);        // K 16..23 / 24..31
  return u.v;
}

// B fragment (32x16 bf16): column n=lane&15 staged K-contiguous in LDS,
// lanes 0-15 hold K=0..7,16..23; lanes 16-31 hold K=8..15,24..31.
DEVINL bf16x16 fragB(const unsigned int* s, int colBase, int lane) {
  int n  = lane & 15;
  int kh = (lane >> 4) << 2;              // dword offset: 0 or 4
  const unsigned int* p = s + (colBase + n) * LDS_PITCH_U32 + kh;
  union { u32x4 q[2]; bf16x16 v; } u;
  u.q[0] = *(const u32x4*)(p);
  u.q[1] = *(const u32x4*)(p + 8);
  return u.v;
}

// ---------------- generic batched GEMM ----------------
// C[z][m][n] = act(alpha * sum_k A[z][m][k]*B[z][k][n] + bias[n] + resid[m][n])
// z = b*nh + h ; per-z bases via batch/head strides. bTrans: B stored [N][K].

struct GemmArgs {
  const float* A; const float* B; float* C;
  const float* bias; const float* resid;
  long long lda, ldb, ldc, ldres;
  long long sAb, sAh, sBb, sBh, sCb, sCh;
  int M, N, K, nh, bTrans, act;   // act: 0=none 1=tanh 2=gelu(tanh approx)
  float alpha;
};

__global__ __launch_bounds__(256) void gemm_bf16_k(GemmArgs g) {
  __shared__ unsigned int sA[2][64  * LDS_PITCH_U32];   // 2 x  5.0 KB
  __shared__ unsigned int sB[2][128 * LDS_PITCH_U32];   // 2 x 10.0 KB

  const int tid   = threadIdx.x;
  const int lane  = tid & 31;
  const int wave  = tid >> 5;
  const int waveM = wave >> 2;          // 0..1
  const int waveN = wave & 3;           // 0..3
  const int blockM = blockIdx.y * 64;
  const int blockN = blockIdx.x * 128;

  const int z  = blockIdx.z;
  const int bb = z / g.nh;
  const int hh = z - bb * g.nh;
  const float* A = g.A + bb * g.sAb + hh * g.sAh;
  const float* B = g.B + bb * g.sBb + hh * g.sBh;
  float*       C = g.C + bb * g.sCb + hh * g.sCh;

  // --- branchless staging coordinates (addresses clamped, never guarded) ---
  // A tile: 64x32, thread -> row ar, 8 consecutive K at column ac.
  const int ar = tid >> 2;                  // 0..63
  const int ac = (tid & 3) << 3;            // 0,8,16,24
  const float* aBase = A + (long long)(blockM + ar) * g.lda + ac;

  // B tile, normal [K,N]: thread -> K-row kR, 16 consecutive N at nC.
  const int kR  = tid >> 3;                 // 0..31
  const int nC  = (tid & 7) << 4;           // 0..112
  int nClamp = blockN + nC;                 // clamp so 16-wide load stays in range;
  if (nClamp > g.N - 16) nClamp = g.N - 16; // garbage cols never stored (N>=16 always)
  const float* bBaseN = B + (long long)kR * g.ldb + nClamp;

  // B tile, transposed [N,K]: thread -> N-row nR, 16 consecutive K at cT.
  const int nR = tid >> 1;                  // 0..127
  const int cT = (tid & 1) << 4;            // 0 or 16
  int nRc = blockN + nR;
  if (nRc > g.N - 1) nRc = g.N - 1;
  const float* bBaseT = B + (long long)nRc * g.ldb + cT;

  float vA[8];
  float vB[16];

  auto loadA = [&](int k0) {
    const float4 a0 = *(const float4*)(aBase + k0);
    const float4 a1 = *(const float4*)(aBase + k0 + 4);
    vA[0]=a0.x; vA[1]=a0.y; vA[2]=a0.z; vA[3]=a0.w;
    vA[4]=a1.x; vA[5]=a1.y; vA[6]=a1.z; vA[7]=a1.w;
  };
  auto loadB = [&](int k0) {
    if (!g.bTrans) {
      const float* bp = bBaseN + (long long)k0 * g.ldb;
#pragma unroll
      for (int q = 0; q < 4; ++q) {
        float4 b4 = *(const float4*)(bp + q * 4);
        vB[q*4+0]=b4.x; vB[q*4+1]=b4.y; vB[q*4+2]=b4.z; vB[q*4+3]=b4.w;
      }
    } else {
      const float* bp = bBaseT + k0;
#pragma unroll
      for (int q = 0; q < 4; ++q) {
        float4 b4 = *(const float4*)(bp + q * 4);
        vB[q*4+0]=b4.x; vB[q*4+1]=b4.y; vB[q*4+2]=b4.z; vB[q*4+3]=b4.w;
      }
    }
  };
  auto stage = [&](int p) {
    u32x4 pk;
    pk.x = pack2(vA[0], vA[1]); pk.y = pack2(vA[2], vA[3]);
    pk.z = pack2(vA[4], vA[5]); pk.w = pack2(vA[6], vA[7]);
    *(u32x4*)&sA[p][ar * LDS_PITCH_U32 + (ac >> 1)] = pk;
    if (!g.bTrans) {
      // transpose 1x16 row chunk into 16 K-contiguous columns
      unsigned short* sB16 = (unsigned short*)&sB[p][0];
#pragma unroll
      for (int i = 0; i < 16; ++i)
        sB16[(nC + i) * (2 * LDS_PITCH_U32) + kR] = f2bf(vB[i]);
    } else {
      u32x4 p0, p1;
      p0.x = pack2(vB[0], vB[1]);   p0.y = pack2(vB[2], vB[3]);
      p0.z = pack2(vB[4], vB[5]);   p0.w = pack2(vB[6], vB[7]);
      p1.x = pack2(vB[8], vB[9]);   p1.y = pack2(vB[10], vB[11]);
      p1.z = pack2(vB[12], vB[13]); p1.w = pack2(vB[14], vB[15]);
      *(u32x4*)&sB[p][nR * LDS_PITCH_U32 + (cT >> 1)]     = p0;
      *(u32x4*)&sB[p][nR * LDS_PITCH_U32 + (cT >> 1) + 4] = p1;
    }
  };

  f32x8 zero = {};
  f32x8 acc[2][2];
  acc[0][0] = zero; acc[0][1] = zero; acc[1][0] = zero; acc[1][1] = zero;

  loadA(0);
  loadB(0);
  int p = 0;
  for (int k0 = 0; k0 < g.K; k0 += 32) {
    stage(p);
    __syncthreads();
    if (k0 + 32 < g.K) {            // prefetch next tile into registers
      loadA(k0 + 32);
      loadB(k0 + 32);
    }
    if (k0 + 64 < g.K) {            // pull tile-after-next toward near caches
      __builtin_prefetch(aBase + k0 + 64, 0, 3);
      __builtin_prefetch((g.bTrans ? bBaseT + k0 + 64
                                   : bBaseN + (long long)(k0 + 64) * g.ldb), 0, 3);
    }
    bf16x16 a0 = fragA(&sA[p][0], waveM * 32,      lane);
    bf16x16 a1 = fragA(&sA[p][0], waveM * 32 + 16, lane);
    bf16x16 b0 = fragB(&sB[p][0], waveN * 32,      lane);
    bf16x16 b1 = fragB(&sB[p][0], waveN * 32 + 16, lane);
    acc[0][0] = wmma_bf16(a0, b0, acc[0][0]);
    acc[0][1] = wmma_bf16(a0, b1, acc[0][1]);
    acc[1][0] = wmma_bf16(a1, b0, acc[1][0]);
    acc[1][1] = wmma_bf16(a1, b1, acc[1][1]);
    p ^= 1;
  }

  // ---- epilogue: C/D layout (ISA 7.12.2): n=lane&15, m=v+8*(lane>>4) ----
  const int laneHi = lane >> 4;
  const int nLane  = lane & 15;
#pragma unroll
  for (int tm = 0; tm < 2; ++tm) {
#pragma unroll
    for (int tn = 0; tn < 2; ++tn) {
      f32x8 a = acc[tm][tn];
      int nG = blockN + waveN * 32 + tn * 16 + nLane;
      if (nG < g.N) {               // M % 64 == 0: no row guard needed
#pragma unroll
        for (int v = 0; v < 8; ++v) {
          int mG = blockM + waveM * 32 + tm * 16 + v + laneHi * 8;
          float val = a[v] * g.alpha;
          if (g.bias)  val += g.bias[nG];
          if (g.resid) val += g.resid[(long long)mG * g.ldres + nG];
          if (g.act == 1) {
            val = tanhf(val);
          } else if (g.act == 2) {
            float x = val;
            val = 0.5f * x * (1.f + tanhf(0.7978845608028654f * (x + 0.044715f * x * x * x)));
          }
          C[(long long)mG * g.ldc + nG] = val;
        }
      }
    }
  }
}

// ---------------- elementwise / reduction kernels ----------------

__global__ void patchify_k(const float* __restrict__ img, float* __restrict__ P,
                           long long total) {
  long long i = (long long)blockIdx.x * blockDim.x + threadIdx.x;
  if (i >= total) return;
  int p = (int)(i & 4095);          // within-patch 16^3
  int s = (int)((i >> 12) & 511);   // token 8^3
  int b = (int)(i >> 21);
  int p3 = p & 15, p2 = (p >> 4) & 15, p1 = p >> 8;
  int g3 = s & 7,  g2 = (s >> 3) & 7, g1 = s >> 6;
  long long src = (((long long)b * 128 + (g1 * 16 + p1)) * 128 + (g2 * 16 + p2)) * 128
                  + (g3 * 16 + p3);
  P[i] = img[src];
}

__global__ void addpos_k(float* __restrict__ x, const float* __restrict__ pos,
                         int total, int SH) {
  int i = blockIdx.x * 256 + threadIdx.x;
  if (i < total) x[i] += pos[i % SH];
}

__global__ __launch_bounds__(256) void layernorm_k(const float* __restrict__ x,
    const float* __restrict__ w, const float* __restrict__ b,
    float* __restrict__ out, int H) {
  __shared__ float red[8];
  __shared__ float stat[2];
  long long row = blockIdx.x;
  const float* xr = x + row * H;
  int tid = threadIdx.x;

  float s = 0.f;
  for (int c = tid; c < H; c += 256) s += xr[c];
  for (int o = 16; o > 0; o >>= 1) s += __shfl_down(s, o, 32);
  if ((tid & 31) == 0) red[tid >> 5] = s;
  __syncthreads();
  if (tid == 0) { float t = 0.f; for (int i = 0; i < 8; ++i) t += red[i]; stat[0] = t / H; }
  __syncthreads();
  float mu = stat[0];

  float v = 0.f;
  for (int c = tid; c < H; c += 256) { float d = xr[c] - mu; v += d * d; }
  for (int o = 16; o > 0; o >>= 1) v += __shfl_down(v, o, 32);
  __syncthreads();
  if ((tid & 31) == 0) red[tid >> 5] = v;
  __syncthreads();
  if (tid == 0) { float t = 0.f; for (int i = 0; i < 8; ++i) t += red[i];
                  stat[1] = rsqrtf(t / H + 1e-5f); }
  __syncthreads();
  float rstd = stat[1];
  for (int c = tid; c < H; c += 256)
    out[row * H + c] = (xr[c] - mu) * rstd * w[c] + b[c];
}

__global__ __launch_bounds__(256) void softmax512_k(float* __restrict__ sc) {
  __shared__ float red[8];
  long long row = blockIdx.x;
  float* r = sc + row * 512;
  int tid = threadIdx.x;
  float a = r[tid], b2 = r[tid + 256];

  float m = fmaxf(a, b2);
  for (int o = 16; o > 0; o >>= 1) m = fmaxf(m, __shfl_down(m, o, 32));
  if ((tid & 31) == 0) red[tid >> 5] = m;
  __syncthreads();
  if (tid == 0) { float t = red[0]; for (int i = 1; i < 8; ++i) t = fmaxf(t, red[i]); red[0] = t; }
  __syncthreads();
  m = red[0];
  float ea = __expf(a - m), eb = __expf(b2 - m);
  float s = ea + eb;
  for (int o = 16; o > 0; o >>= 1) s += __shfl_down(s, o, 32);
  __syncthreads();
  if ((tid & 31) == 0) red[tid >> 5] = s;
  __syncthreads();
  if (tid == 0) { float t = 0.f; for (int i = 0; i < 8; ++i) t += red[i]; red[0] = t; }
  __syncthreads();
  float inv = 1.0f / red[0];
  r[tid] = ea * inv;
  r[tid + 256] = eb * inv;
}

__global__ void zero2_k(float* p) { if (threadIdx.x < 2) p[threadIdx.x] = 0.f; }

// Fused VQ: l2-normalize feat row, cosine argmax over 8192 codes,
// gather code, commitment losses (both equal in forward), indices as floats.
__global__ __launch_bounds__(256) void vq_k(const float* __restrict__ feat,
    const float* __restrict__ codebook, float* __restrict__ code,
    float* __restrict__ lossOut, float* __restrict__ idxOut) {
  __shared__ float f[32];
  __shared__ float bestV[8];
  __shared__ int   bestI[8];
  int t = blockIdx.x;                     // token 0..4095
  if (threadIdx.x < 32) f[threadIdx.x] = feat[t * 32 + threadIdx.x];
  __syncthreads();

  float nrm = 0.f;
#pragma unroll
  for (int d = 0; d < 32; ++d) nrm += f[d] * f[d];
  float inv = 1.0f / fmaxf(sqrtf(nrm), 1e-12f);

  float bv = -3.4e38f; int bi = 0;
  for (int c = threadIdx.x; c < 8192; c += 256) {
    const float4* cb = (const float4*)(codebook + c * 32);
    const float4* fr = (const float4*)f;
    float dot = 0.f;
#pragma unroll
    for (int q = 0; q < 8; ++q) {
      float4 cc = cb[q]; float4 ff = fr[q];
      dot += cc.x * ff.x + cc.y * ff.y + cc.z * ff.z + cc.w * ff.w;
    }
    dot *= inv;
    if (dot > bv) { bv = dot; bi = c; }
  }
  for (int o = 16; o > 0; o >>= 1) {
    float ov = __shfl_down(bv, o, 32);
    int   oi = __shfl_down(bi, o, 32);
    if (ov > bv || (ov == bv && oi < bi)) { bv = ov; bi = oi; }
  }
  if ((threadIdx.x & 31) == 0) { bestV[threadIdx.x >> 5] = bv; bestI[threadIdx.x >> 5] = bi; }
  __syncthreads();
  if (threadIdx.x == 0) {
    for (int i = 1; i < 8; ++i)
      if (bestV[i] > bv || (bestV[i] == bv && bestI[i] < bi)) { bv = bestV[i]; bi = bestI[i]; }
    bestI[0] = bi;
    idxOut[t] = (float)bi;
  }
  __syncthreads();
  bi = bestI[0];
  if (threadIdx.x < 32) {
    float cd = codebook[bi * 32 + threadIdx.x];
    code[t * 32 + threadIdx.x] = cd;            // straight-through forward = code
    float d = cd - f[threadIdx.x];
    float c2 = d * d * (1.0f / 131072.0f);      // mean over 8*512*32
    for (int o = 16; o > 0; o >>= 1) c2 += __shfl_down(c2, o, 32);
    if (threadIdx.x == 0) { atomicAdd(&lossOut[0], c2); atomicAdd(&lossOut[1], c2); }
  }
}

// ---------------- host orchestration ----------------

static void gemm_launch(hipStream_t st, int M, int N, int K,
    const float* A, long long lda,
    const float* B, long long ldb, int bTrans,
    float* C, long long ldc,
    const float* bias, const float* resid, long long ldres,
    float alpha, int act,
    int nBatch = 1, int nHead = 1,
    long long sAb = 0, long long sAh = 0,
    long long sBb = 0, long long sBh = 0,
    long long sCb = 0, long long sCh = 0) {
  GemmArgs g;
  g.A = A; g.B = B; g.C = C; g.bias = bias; g.resid = resid;
  g.lda = lda; g.ldb = ldb; g.ldc = ldc; g.ldres = ldres;
  g.sAb = sAb; g.sAh = sAh; g.sBb = sBb; g.sBh = sBh; g.sCb = sCb; g.sCh = sCh;
  g.M = M; g.N = N; g.K = K; g.nh = nHead; g.bTrans = bTrans; g.act = act; g.alpha = alpha;
  dim3 grid((N + 127) / 128, (M + 63) / 64, nBatch * nHead);
  gemm_bf16_k<<<grid, dim3(256), 0, st>>>(g);
}

struct VitW {
  const float *ln1w, *ln1b, *qkvw, *qkvb, *ow, *ob;
  const float *ln2w, *ln2b, *fc1w, *fc1b, *fc2w, *fc2b;
};

// X: residual stream [4096,768]; T: scratch [4096,768]; QKV: [4096,2304];
// SC: big region (scores [96,512,512] / MLP hidden [4096,3072], time-shared).
static void run_vit(hipStream_t st, float* X, float* T, float* QKV, float* SC,
                    const VitW& w, int L) {
  for (int l = 0; l < L; ++l) {
    const float* ln1w = w.ln1w + (long long)l * 768;
    const float* ln1b = w.ln1b + (long long)l * 768;
    const float* qkvw = w.qkvw + (long long)l * 768 * 2304;
    const float* qkvb = w.qkvb + (long long)l * 2304;
    const float* ow   = w.ow   + (long long)l * 768 * 768;
    const float* ob   = w.ob   + (long long)l * 768;
    const float* ln2w = w.ln2w + (long long)l * 768;
    const float* ln2b = w.ln2b + (long long)l * 768;
    const float* fc1w = w.fc1w + (long long)l * 768 * 3072;
    const float* fc1b = w.fc1b + (long long)l * 3072;
    const float* fc2w = w.fc2w + (long long)l * 3072 * 768;
    const float* fc2b = w.fc2b + (long long)l * 768;

    layernorm_k<<<4096, 256, 0, st>>>(X, ln1w, ln1b, T, 768);
    gemm_launch(st, 4096, 2304, 768, T, 768, qkvw, 2304, 0, QKV, 2304,
                qkvb, nullptr, 0, 1.f, 0);
    // scores[z] = (Q @ K^T) * hd^-0.5 ; z = b*12+h (96 batched GEMMs)
    gemm_launch(st, 512, 512, 64,
                QKV, 2304, QKV + 768, 2304, /*bTrans=*/1, SC, 512,
                nullptr, nullptr, 0, 0.125f, 0,
                8, 12,
                512LL * 2304, 64,
                512LL * 2304, 64,
                12LL * 512 * 512, 512LL * 512);
    softmax512_k<<<96 * 512, 256, 0, st>>>(SC);
    // T[heads] = att @ V
    gemm_launch(st, 512, 64, 512,
                SC, 512, QKV + 1536, 2304, 0, T, 768,
                nullptr, nullptr, 0, 1.f, 0,
                8, 12,
                12LL * 512 * 512, 512LL * 512,
                512LL * 2304, 64,
                512LL * 768, 64);
    // X = X + T @ o_w + o_b
    gemm_launch(st, 4096, 768, 768, T, 768, ow, 768, 0, X, 768,
                ob, X, 768, 1.f, 0);
    layernorm_k<<<4096, 256, 0, st>>>(X, ln2w, ln2b, T, 768);
    // SC = gelu(T @ fc1 + b1)
    gemm_launch(st, 4096, 3072, 768, T, 768, fc1w, 3072, 0, SC, 3072,
                fc1b, nullptr, 0, 1.f, 2);
    // X = X + SC @ fc2 + b2
    gemm_launch(st, 4096, 768, 3072, SC, 3072, fc2w, 768, 0, X, 768,
                fc2b, X, 768, 1.f, 0);
  }
}

// d_in order: recursive insertion order of setup_inputs()
enum {
  IN_IMG = 0, IN_ENC_PE_W, IN_ENC_PE_B, IN_ENC_POS,
  IN_E_LN1W, IN_E_LN1B, IN_E_QKVW, IN_E_QKVB, IN_E_OW, IN_E_OB,
  IN_E_LN2W, IN_E_LN2B, IN_E_FC1W, IN_E_FC1B, IN_E_FC2W, IN_E_FC2B,
  IN_ENC_LNF_W, IN_ENC_LNF_B,
  IN_DEC_PE_W, IN_DEC_PE_B, IN_DEC_POS,
  IN_D_LN1W, IN_D_LN1B, IN_D_QKVW, IN_D_QKVB, IN_D_OW, IN_D_OB,
  IN_D_LN2W, IN_D_LN2B, IN_D_FC1W, IN_D_FC1B, IN_D_FC2W, IN_D_FC2B,
  IN_DEC_LNF_W, IN_DEC_LNF_B,
  IN_IN_W1, IN_IN_B1, IN_IN_W2, IN_IN_B2,
  IN_OUT_W1, IN_OUT_B1, IN_OUT_W2, IN_OUT_B2,
  IN_VQ
};

extern "C" void kernel_launch(void* const* d_in, const int* in_sizes, int n_in,
                              void* d_out, int out_size, void* d_ws, size_t ws_size,
                              hipStream_t stream) {
  (void)in_sizes; (void)n_in; (void)out_size; (void)ws_size;
  #define INF(i) ((const float*)d_in[(i)])

  // Workspace arena (floats), ~157 MB total:
  //   BIG: 25,165,824 (patches 16.8M / scores 25.2M / mlp-hidden 12.6M, time-shared)
  //   X,T: 3,145,728 each ; QKV: 9,437,184 ; FEAT/CODE: 131,072 each
  float* ws   = (float*)d_ws;
  float* BIG  = ws;
  float* X    = BIG  + 25165824LL;
  float* T    = X    + 3145728LL;
  float* QKV  = T    + 3145728LL;
  float* FEAT = QKV  + 9437184LL;
  float* CODE = FEAT + 131072LL;

  // ---- encoder ----
  long long totalP = 8LL * 512 * 4096;
  patchify_k<<<(unsigned)((totalP + 255) / 256), 256, 0, stream>>>(INF(IN_IMG), BIG, totalP);
  gemm_launch(stream, 4096, 768, 4096, BIG, 4096, INF(IN_ENC_PE_W), 768, 0,
              X, 768, INF(IN_ENC_PE_B), nullptr, 0, 1.f, 0);
  addpos_k<<<12288, 256, 0, stream>>>(X, INF(IN_ENC_POS), 3145728, 393216);

  VitW encW = { INF(IN_E_LN1W), INF(IN_E_LN1B), INF(IN_E_QKVW), INF(IN_E_QKVB),
                INF(IN_E_OW),   INF(IN_E_OB),   INF(IN_E_LN2W), INF(IN_E_LN2B),
                INF(IN_E_FC1W), INF(IN_E_FC1B), INF(IN_E_FC2W), INF(IN_E_FC2B) };
  run_vit(stream, X, T, QKV, BIG, encW, 12);

  layernorm_k<<<4096, 256, 0, stream>>>(X, INF(IN_ENC_LNF_W), INF(IN_ENC_LNF_B), T, 768);

  // input_mlp: feat = tanh(T @ in_w1 + b1) @ in_w2 + b2  (QKV reused as hidden)
  gemm_launch(stream, 4096, 768, 768, T, 768, INF(IN_IN_W1), 768, 0, QKV, 768,
              INF(IN_IN_B1), nullptr, 0, 1.f, 1);
  gemm_launch(stream, 4096, 32, 768, QKV, 768, INF(IN_IN_W2), 32, 0, FEAT, 32,
              INF(IN_IN_B2), nullptr, 0, 1.f, 0);

  // ---- VQ ----
  float* outF    = (float*)d_out;
  float* lossPtr = outF + 3145728;       // [l_encoder2embed, l_embed2encoder]
  float* idxPtr  = outF + 3145730;       // 4096 indices (as floats)
  zero2_k<<<1, 32, 0, stream>>>(lossPtr);
  vq_k<<<4096, 256, 0, stream>>>(FEAT, INF(IN_VQ), CODE, lossPtr, idxPtr);

  // ---- decoder (bridge == code in forward) ----
  gemm_launch(stream, 4096, 768, 32, CODE, 32, INF(IN_DEC_PE_W), 768, 0,
              X, 768, INF(IN_DEC_PE_B), nullptr, 0, 1.f, 0);
  addpos_k<<<12288, 256, 0, stream>>>(X, INF(IN_DEC_POS), 3145728, 393216);

  VitW decW = { INF(IN_D_LN1W), INF(IN_D_LN1B), INF(IN_D_QKVW), INF(IN_D_QKVB),
                INF(IN_D_OW),   INF(IN_D_OB),   INF(IN_D_LN2W), INF(IN_D_LN2B),
                INF(IN_D_FC1W), INF(IN_D_FC1B), INF(IN_D_FC2W), INF(IN_D_FC2B) };
  run_vit(stream, X, T, QKV, BIG, decW, 3);

  layernorm_k<<<4096, 256, 0, stream>>>(X, INF(IN_DEC_LNF_W), INF(IN_DEC_LNF_B), T, 768);

  // output_mlp -> d_out[0 : 4096*768]
  gemm_launch(stream, 4096, 768, 768, T, 768, INF(IN_OUT_W1), 768, 0, QKV, 768,
              INF(IN_OUT_B1), nullptr, 0, 1.f, 1);
  gemm_launch(stream, 4096, 768, 768, QKV, 768, INF(IN_OUT_W2), 768, 0, outF, 768,
              INF(IN_OUT_B2), nullptr, 0, 1.f, 0);
  #undef INF
}
